// MultiHeadConvAttentionBlock_58755152609399
// MI455X (gfx1250) — matrix operand order repaired and checked by reading, not compiled, not run on hardware
//
#include <hip/hip_runtime.h>

typedef __bf16 bf16;
typedef __attribute__((ext_vector_type(16))) __bf16 v16bf;
typedef __attribute__((ext_vector_type(8)))  float  v8f;

#define NB   4
#define NL   1024
#define NH   8
#define NDE  64
#define NCIN 256
#define NCO  512
#define NKIC 768      // im2col K = 3*256
#define NPOS 65
#define MAXPOS 32

// D = A(16x32 bf16) * B(32x16 bf16) + C(16x16 f32)   -> v_wmma_f32_16x16x32_bf16
__device__ __forceinline__ v8f wmma_bf(v16bf a, v16bf b, v8f c) {
  return __builtin_amdgcn_wmma_f32_16x16x32_bf16(false, a, false, b, (short)0, c, false, false);
}

// B-matrix tile (32x16 bf16): lane -> N = lane%16, elements = K in table order (row stride ld)
__device__ __forceinline__ v16bf load_b16(const bf16* p, int ld) {
  int ln = threadIdx.x & 31;
  int n = ln & 15, hh = ln >> 4;
  v16bf b;
#pragma unroll
  for (int s = 0; s < 16; ++s) {
    int k = (s < 8) ? (8 * hh + s) : (16 + 8 * hh + (s - 8));
    b[s] = p[(size_t)k * ld + n];
  }
  return b;
}

// A-matrix tile (16x32 bf16) from a contiguous 32-element row:
// per-lane elements = [half*8 .. half*8+7] and [16+half*8 .. 16+half*8+7]
__device__ __forceinline__ v16bf load_a16(const bf16* p, int half) {
  union { uint4 u[2]; v16bf b; } c;
  c.u[0] = *(const uint4*)(p + half * 8);
  c.u[1] = *(const uint4*)(p + 16 + half * 8);
  return c.b;
}

// Build A-tile register from 16 floats (two contiguous runs of 8)
__device__ __forceinline__ v16bf pack_a16(float4 f0, float4 f1, float4 f2, float4 f3) {
  v16bf a;
  a[0]  = (bf16)f0.x; a[1]  = (bf16)f0.y; a[2]  = (bf16)f0.z; a[3]  = (bf16)f0.w;
  a[4]  = (bf16)f1.x; a[5]  = (bf16)f1.y; a[6]  = (bf16)f1.z; a[7]  = (bf16)f1.w;
  a[8]  = (bf16)f2.x; a[9]  = (bf16)f2.y; a[10] = (bf16)f2.z; a[11] = (bf16)f2.w;
  a[12] = (bf16)f3.x; a[13] = (bf16)f3.y; a[14] = (bf16)f3.z; a[15] = (bf16)f3.w;
  return a;
}

// ---------------------------------------------------------------------------
// Weight prep: im2col-transpose conv weights to bf16 [k=(ci+256*t)][co],
// transpose Wfc to bf16 [m][n].
// ---------------------------------------------------------------------------
__global__ void prep_weights(const float* __restrict__ Wq, const float* __restrict__ Wk,
                             const float* __restrict__ Wv, const float* __restrict__ Wfc,
                             bf16* __restrict__ Wqt, bf16* __restrict__ Wkt,
                             bf16* __restrict__ Wvt, bf16* __restrict__ Wfct) {
  int tid = blockIdx.x * blockDim.x + threadIdx.x;
  const int NW = NKIC * NCO;                 // 393216 per conv weight
  if (tid < 3 * NW) {
    int which = tid / NW;
    int idx   = tid % NW;
    int kk = idx / NCO, co = idx % NCO;
    int ci = kk & 255, t = kk >> 8;
    const float* src = (which == 0) ? Wq : (which == 1) ? Wk : Wv;
    bf16*        dst = (which == 0) ? Wqt : (which == 1) ? Wkt : Wvt;
    dst[idx] = (bf16)src[(co * NCIN + ci) * 3 + t];
  } else {
    int idx = tid - 3 * NW;
    if (idx < NCO * NCO) {
      int m = idx / NCO, n = idx % NCO;
      Wfct[idx] = (bf16)Wfc[n * NCO + m];    // [m][n]
    }
  }
}

// ---------------------------------------------------------------------------
// Conv1d projection as im2col GEMM. One wave -> 16(l) x 16(co) C tile,
// K=768 in 24 bf16 WMMAs.  Chunk size 32 divides 256, so the conv tap
// t = kc>>3 (and thus the padding bounds check) is uniform per chunk.
// dmajor==0: write [b][h][l][d] (Q, V); dmajor==1: write [b][h][d][l] (K).
// ---------------------------------------------------------------------------
__global__ void __launch_bounds__(128)
conv_proj(const float* __restrict__ x, const bf16* __restrict__ Wt,
          const float* __restrict__ bias, bf16* __restrict__ out, int dmajor) {
  int wid   = blockIdx.x * (blockDim.x >> 5) + (threadIdx.x >> 5);
  int lane  = threadIdx.x & 31;
  int cotile = wid & 31;
  int ltile  = (wid >> 5) & 63;
  int b      = wid >> 11;
  int half = lane >> 4;
  int mrow = lane & 15;
  v8f acc = {};
  for (int kc = 0; kc < 24; ++kc) {
    int t       = kc >> 3;              // conv tap, uniform per chunk
    int base_ci = (kc & 7) * 32;        // input-channel base, uniform per chunk
    int lrow    = ltile * 16 + mrow - 1 + t;
    float4 f0 = {}, f1 = {}, f2 = {}, f3 = {};
    if (lrow >= 0 && lrow < NL) {       // single hoisted pad check per chunk
      const float* p = x + ((size_t)b * NL + lrow) * NCIN + base_ci + half * 8;
      f0 = *(const float4*)(p);
      f1 = *(const float4*)(p + 4);
      f2 = *(const float4*)(p + 16);
      f3 = *(const float4*)(p + 20);
    }
    v16bf aA = pack_a16(f0, f1, f2, f3);
    v16bf bB = load_b16(Wt + ((size_t)kc * 32) * NCO + cotile * 16, NCO);
    acc = wmma_bf(aA, bB, acc);
  }
#pragma unroll
  for (int i = 0; i < 8; ++i) {                     // C: VGPR i -> M = i + 8*half
    int m  = i + 8 * half;
    int l  = ltile * 16 + m;
    int co = cotile * 16 + mrow;
    float val = acc[i] + bias[co];
    int h = co >> 6, d = co & 63;
    size_t off = dmajor ? ((((size_t)b * NH + h) * NDE + d) * NL + l)
                        : ((((size_t)b * NH + h) * NL + l) * NDE + d);
    out[off] = (bf16)val;
  }
}

// ---------------------------------------------------------------------------
// Relative-position factorization: pos[b,h,q,r] = sum_d Q[b,h,q,d] * w_pos[r,d]
// ---------------------------------------------------------------------------
__global__ void pos_proj(const bf16* __restrict__ Qbf, const float* __restrict__ w_pos,
                         float* __restrict__ pos) {
  int tid = blockIdx.x * blockDim.x + threadIdx.x;
  if (tid >= NB * NH * NL * NPOS) return;
  int r   = tid % NPOS;
  int bhq = tid / NPOS;
  const bf16*  qp = Qbf + (size_t)bhq * NDE;
  const float* wp = w_pos + r * NDE;
  float acc = 0.f;
#pragma unroll 8
  for (int d = 0; d < NDE; ++d) acc += (float)qp[d] * wp[d];
  pos[tid] = acc;
}

// ---------------------------------------------------------------------------
// Fused attention: one wave per (b, h, 16-row q-tile).
// Pass1: S = QK^T/8 + pos, mask -> LDS [16][1024]
// Pass2: softmax (shfl_xor across half-waves), A*(A>vm), write A (float4)
// Pass3: O = A*V via 4 WMMA accumulators (d = 64), store bf16
// ---------------------------------------------------------------------------
__global__ void __launch_bounds__(32)
attn_kernel(const bf16* __restrict__ Qbf, const bf16* __restrict__ Kd,
            const bf16* __restrict__ Vbf, const float* __restrict__ pos,
            const unsigned char* __restrict__ mask, const float* __restrict__ valid_mask,
            float* __restrict__ Aout, bf16* __restrict__ Obf) {
  __shared__ float S[16 * NL];                       // 64 KB of the 320 KB WGP LDS
  int lane = threadIdx.x & 31;
  int half = lane >> 4;
  int mrow = lane & 15;
  int qtile = blockIdx.x & 63;
  int h     = (blockIdx.x >> 6) & 7;
  int b     = blockIdx.x >> 9;
  int qbase = qtile * 16;
  size_t bh = (size_t)b * NH + h;
  const float NEG_INF = -__builtin_inff();

  // Q A-tiles (reused across all 64 k-tiles): d = 0..31 and 32..63
  const bf16* qp = Qbf + (bh * NL + qbase + mrow) * NDE;
  v16bf aQ0 = load_a16(qp, half);
  v16bf aQ1 = load_a16(qp + 32, half);

  // -------- Pass 1: scores into LDS --------
  for (int kt = 0; kt < 64; ++kt) {
    int kbase = kt * 16;
    v16bf bK0 = load_b16(Kd + (bh * NDE)      * NL + kbase, NL);   // d-major K
    v16bf bK1 = load_b16(Kd + (bh * NDE + 32) * NL + kbase, NL);
    v8f c = {};
    c = wmma_bf(aQ0, bK0, c);
    c = wmma_bf(aQ1, bK1, c);
#pragma unroll
    for (int i = 0; i < 8; ++i) {
      int m = i + 8 * half;
      int q = qbase + m;
      int k = kbase + mrow;
      int rel = k - q;
      rel = rel < -MAXPOS ? -MAXPOS : (rel > MAXPOS ? MAXPOS : rel);
      float s = c[i] * 0.125f + pos[(bh * NL + q) * NPOS + rel + MAXPOS];
      if (mask[(size_t)q * NL + k]) s = NEG_INF;
      S[m * NL + k] = s;
    }
  }
  __syncthreads();

  // -------- Pass 2: softmax + A*(A>vm) --------
  float* rowp = S + mrow * NL + half * 512;          // lane scans half a row
  float mx = NEG_INF;
  for (int j = 0; j < 512; ++j) mx = fmaxf(mx, rowp[j]);
  mx = fmaxf(mx, __shfl_xor(mx, 16, 32));
  float sum = 0.f;
  for (int j = 0; j < 512; ++j) {
    float s = rowp[j];
    float e = (s == NEG_INF) ? 0.f : __expf(s - mx);
    rowp[j] = e;
    sum += e;
  }
  sum += __shfl_xor(sum, 16, 32);
  float inv = 1.f / sum;
  float vm  = valid_mask[0];
  for (int j = 0; j < 512; ++j) {
    float a = rowp[j] * inv;
    rowp[j] = (a > vm) ? a : 0.f;
  }
  __syncthreads();

  // -------- Write A (coalesced float4) --------
  {
    float* ap = Aout + (bh * NL + qbase) * NL;       // same [16][1024] layout as S
    for (int i = lane * 4; i < 16 * NL; i += 32 * 4)
      *(float4*)(ap + i) = *(const float4*)(S + i);
  }

  // -------- Pass 3: O = A * V --------
  v8f acc[4] = {{}, {}, {}, {}};
  for (int kc = 0; kc < 32; ++kc) {
    const float* sp = S + mrow * NL + kc * 32 + half * 8;   // 32B-aligned
    float4 s0 = *(const float4*)(sp);
    float4 s1 = *(const float4*)(sp + 4);
    float4 s2 = *(const float4*)(sp + 16);
    float4 s3 = *(const float4*)(sp + 20);
    v16bf aA = pack_a16(s0, s1, s2, s3);
    const bf16* vp = Vbf + (bh * NL + kc * 32) * NDE;
#pragma unroll
    for (int ng = 0; ng < 4; ++ng) {
      v16bf bV = load_b16(vp + ng * 16, NDE);
      acc[ng] = wmma_bf(aA, bV, acc[ng]);
    }
  }
  bf16* op = Obf + ((size_t)b * NL + qbase) * NCO + h * NDE;   // O -> [b][l][h*64+d]
#pragma unroll
  for (int ng = 0; ng < 4; ++ng)
#pragma unroll
    for (int i = 0; i < 8; ++i) {
      int m = i + 8 * half;
      op[(size_t)m * NCO + ng * 16 + mrow] = (bf16)acc[ng][i];
    }
}

// ---------------------------------------------------------------------------
// FC: out[bl][n] = O[bl][:] . Wfc[n][:] + bfc[n]  (M=4096, N=512, K=512)
// ---------------------------------------------------------------------------
__global__ void __launch_bounds__(128)
fc_kernel(const bf16* __restrict__ Obf, const bf16* __restrict__ Wfct,
          const float* __restrict__ bfc, float* __restrict__ out) {
  int wid  = blockIdx.x * (blockDim.x >> 5) + (threadIdx.x >> 5);
  int lane = threadIdx.x & 31;
  int ntile = wid & 31;
  int mtile = wid >> 5;
  int half = lane >> 4, mrow = lane & 15;
  v8f acc = {};
  for (int kc = 0; kc < 16; ++kc) {
    v16bf aA = load_a16(Obf + ((size_t)mtile * 16 + mrow) * NCO + kc * 32, half);
    v16bf bB = load_b16(Wfct + ((size_t)kc * 32) * NCO + ntile * 16, NCO);
    acc = wmma_bf(aA, bB, acc);
  }
#pragma unroll
  for (int i = 0; i < 8; ++i) {
    int m = i + 8 * half;
    int row = mtile * 16 + m;
    int col = ntile * 16 + mrow;
    out[(size_t)row * NCO + col] = acc[i] + bfc[col];
  }
}

// ---------------------------------------------------------------------------
extern "C" void kernel_launch(void* const* d_in, const int* in_sizes, int n_in,
                              void* d_out, int out_size, void* d_ws, size_t ws_size,
                              hipStream_t stream) {
  const float* q    = (const float*)d_in[0];
  const float* k    = (const float*)d_in[1];
  const float* v    = (const float*)d_in[2];
  const float* Wq   = (const float*)d_in[3];
  const float* bq   = (const float*)d_in[4];
  const float* Wk   = (const float*)d_in[5];
  const float* bk   = (const float*)d_in[6];
  const float* Wv   = (const float*)d_in[7];
  const float* bv   = (const float*)d_in[8];
  const float* Wfc  = (const float*)d_in[9];
  const float* bfc  = (const float*)d_in[10];
  const float* wpos = (const float*)d_in[11];
  const unsigned char* attn_mask = (const unsigned char*)d_in[12];
  const float* valid_mask = (const float*)d_in[13];

  float* out  = (float*)d_out;                              // [4,1024,512]
  float* Aout = out + (size_t)NB * NL * NCO;                // [4,8,1024,1024]

  char* ws = (char*)d_ws;
  size_t off = 0;
  auto alloc = [&](size_t bytes) {
    char* p = ws + off;
    off += (bytes + 255) & ~(size_t)255;
    return p;
  };
  bf16* Wqt  = (bf16*)alloc((size_t)NKIC * NCO * 2);
  bf16* Wkt  = (bf16*)alloc((size_t)NKIC * NCO * 2);
  bf16* Wvt  = (bf16*)alloc((size_t)NKIC * NCO * 2);
  bf16* Wfct = (bf16*)alloc((size_t)NCO * NCO * 2);
  bf16* Qbf  = (bf16*)alloc((size_t)NB * NH * NL * NDE * 2);  // [b][h][l][d]
  bf16* Kd   = (bf16*)alloc((size_t)NB * NH * NDE * NL * 2);  // [b][h][d][l]
  bf16* Vbf  = (bf16*)alloc((size_t)NB * NH * NL * NDE * 2);  // [b][h][l][d]
  float* pos = (float*)alloc((size_t)NB * NH * NL * NPOS * 4);
  bf16* Obf  = (bf16*)alloc((size_t)NB * NL * NCO * 2);       // [b*l][512]

  // 1) weight prep
  {
    int total = 3 * NKIC * NCO + NCO * NCO;                   // 1,441,792
    prep_weights<<<(total + 255) / 256, 256, 0, stream>>>(Wq, Wk, Wv, Wfc,
                                                          Wqt, Wkt, Wvt, Wfct);
  }
  // 2) conv projections: 8192 waves each -> 2048 blocks of 128
  conv_proj<<<2048, 128, 0, stream>>>(q, Wqt, bq, Qbf, 0);
  conv_proj<<<2048, 128, 0, stream>>>(k, Wkt, bk, Kd, 1);
  conv_proj<<<2048, 128, 0, stream>>>(v, Wvt, bv, Vbf, 0);
  // 3) relative-position factorization
  {
    int total = NB * NH * NL * NPOS;                          // 2,129,920
    pos_proj<<<(total + 255) / 256, 256, 0, stream>>>(Qbf, wpos, pos);
  }
  // 4) fused attention: one wave per (b,h,qtile)
  attn_kernel<<<NB * NH * 64, 32, 0, stream>>>(Qbf, Kd, Vbf, pos, attn_mask,
                                               valid_mask, Aout, Obf);
  // 5) FC: 8192 waves -> 2048 blocks of 128
  fc_kernel<<<2048, 128, 0, stream>>>(Obf, Wfct, bfc, out);
}
